// CortexGPT_16801912062745
// MI455X (gfx1250) — compile-verified, matrix operand order checked by reading, not run on hardware
//
#include <hip/hip_runtime.h>
#include <hip/hip_bf16.h>

// ---------------------------------------------------------------------------
// CortexGPT forward for MI455X (gfx1250, wave32, WMMA + double-buffered TDM).
// Dominant cost: logits GEMM [2048,768]x[768,50257] -> f16 WMMA w/ f32 accum.
// ---------------------------------------------------------------------------

typedef __attribute__((ext_vector_type(16))) _Float16 v16h;
typedef __attribute__((ext_vector_type(8)))  float    v8f;

#define NTOK 2048
#define DIM  768
#define NCOL 16
#define SEQ  512
#define VOCAB 50257
#define TOPK 102
#define KPAD 112   // 7 tiles of 16
#define APAD 20    // LDS row stride in uints (16 data + 4 pad): 80B rows keep
                   // 16B alignment for b128 LDS ops and conflict-free banks.

// K-pair base for 16-bit A/B WMMA fragments (ISA 7.12.2, 16-bit A 16x32):
// VGPR v in [0,3]: K = 2v + 8g ; VGPR v in [4,7]: K = 16 + 2(v-4) + 8g
__device__ __forceinline__ int kmap(int v, int g) {
    return (v < 4) ? (v * 2 + g * 8) : (16 + (v - 4) * 2 + g * 8);
}

union AFrag { v16h v; uint u[8]; };
union BFrag { v16h v; _Float16 h[16]; };
union HPack { _Float16 h[2]; uint u32; };

// ---------------------------------------------------------------------------
// Tensor Data Mover staging (gfx1250). LDS pad feature reproduces the APAD
// layout in hardware: after every 16 DWORDs (one 32-f16 row) insert 4 DWORDs.
// ---------------------------------------------------------------------------
#if defined(__HIP_DEVICE_COMPILE__) && defined(__gfx1250__) && \
    __has_builtin(__builtin_amdgcn_tensor_load_to_lds)
#define USE_TDM 1
#if __has_include(<hip/amd_detail/amd_gfx1250_TDM.h>)
#define TDM_SIX_ARGS 1
#else
#define TDM_SIX_ARGS 0
#endif
#else
#define USE_TDM 0
#endif

#if USE_TDM
typedef __attribute__((ext_vector_type(4))) unsigned int tdm_v4u;
typedef __attribute__((ext_vector_type(8))) int          tdm_v8i;
typedef __attribute__((ext_vector_type(4))) int          tdm_v4i;

// Load a rows x 32 f16 tile (row-major, leading dim ld_elems) from global to
// LDS at lds_off, with +16B padding per 64B row (APAD layout).
__device__ __forceinline__ void tdm_load_tile(const void* gsrc, uint lds_off,
                                              uint rows, uint ld_elems) {
    unsigned long long ga = (unsigned long long)(uintptr_t)gsrc;
    tdm_v4u g0;
    g0[0] = 1u;                                   // count=1, user desc
    g0[1] = lds_off;                              // LDS byte address
    g0[2] = (uint)ga;                             // global_addr[31:0]
    g0[3] = (uint)((ga >> 32) & 0x01ffffffu)      // global_addr[56:32]
            | (2u << 30);                         // type=2 ("image")
    tdm_v8i g1;
    g1[0] = (int)((1u << 16)      // data_size = 2 bytes
            | (1u << 20)          // pad_enable
            | (3u << 22)          // pad_interval: every 16 DWORDs (64B row)
            | (3u << 25));        // pad_amount: 4 DWORDs (16B)
    g1[1] = (int)((ld_elems & 0xffffu) << 16);            // tensor_dim0 lo16
    g1[2] = (int)(((ld_elems >> 16) & 0xffffu)            // tensor_dim0 hi16
            | ((rows & 0xffffu) << 16));                  // tensor_dim1 lo16
    g1[3] = (int)(((rows >> 16) & 0xffffu)                // tensor_dim1 hi16
            | (32u << 16));                               // tile_dim0 = 32
    g1[4] = (int)(rows & 0xffffu);                        // tile_dim1, dim2=0
    g1[5] = (int)ld_elems;                                // dim0_stride lo32
    g1[6] = 0;                                            // stride hi / d1s lo
    g1[7] = 0;
    tdm_v4i z4 = {0, 0, 0, 0};
#if TDM_SIX_ARGS
    tdm_v8i z8 = {0, 0, 0, 0, 0, 0, 0, 0};
    __builtin_amdgcn_tensor_load_to_lds(g0, g1, z4, z4, z8, 0);
#else
    __builtin_amdgcn_tensor_load_to_lds(g0, g1, z4, z4, 0);
#endif
}
#endif // USE_TDM

// Manual fallback: stage a (MT*16) x 32 f16 tile into LDS (APAD layout).
template <int MT>
__device__ __forceinline__ void stage_A_manual(
    const _Float16* __restrict__ Ag, int lda, int m0, int k0,
    uint* As, int tid) {
    const uint* Agu = (const uint*)Ag;
    int ldau = lda >> 1, k0u = k0 >> 1;
#pragma unroll
    for (int q = tid; q < MT * 64; q += 256) {
        int row = q >> 2, c4 = q & 3;
        const uint4* src =
            (const uint4*)(Agu + (size_t)(m0 + row) * ldau + k0u) + c4;
        *((uint4*)(As + row * APAD) + c4) = *src;
    }
}

// Build B fragment (32x16 f16) from f32 global, one 16-col tile, lane col.
__device__ __forceinline__ v16h load_B_f32(
    const float* __restrict__ Bg, int ldb, int k0, int col, int g) {
    BFrag bf;
#pragma unroll
    for (int v = 0; v < 8; ++v) {
        int kb = kmap(v, g);
        bf.h[2 * v]     = (_Float16)Bg[(size_t)(k0 + kb) * ldb + col];
        bf.h[2 * v + 1] = (_Float16)Bg[(size_t)(k0 + kb + 1) * ldb + col];
    }
    return bf.v;
}

// A fragment for one 16-row tile from staged LDS (two contiguous 16B chunks).
__device__ __forceinline__ v16h load_A_lds(const uint* As, int t, int r, int g) {
    AFrag af;
    const uint* base = As + (t * 16 + r) * APAD + 4 * g;
#pragma unroll
    for (int v = 0; v < 4; ++v) af.u[v] = base[v];
#pragma unroll
    for (int v = 4; v < 8; ++v) af.u[v] = base[4 + v];   // +8 skip, -4 offset
    return af.v;
}

// ---------------------------------------------------------------------------
// Multi-M-tile GEMM body: acc[MT] (16x16 each), one shared B frag per k-step.
// A tile double-buffered in LDS via TDM (issue k+1 before waiting on k);
// B fragment software-pipelined across k-steps. blockDim must be 256.
// As must hold 2 * MT*16*APAD uints.
// ---------------------------------------------------------------------------
template <int MT>
__device__ __forceinline__ void gemm_f16A(
    const _Float16* __restrict__ Ag, int lda, int m0,
    const float* __restrict__ Bg, int ldb, int col,
    int K, uint* As, int tid, v8f* acc) {
    const int lane = tid & 31, r = lane & 15, g = lane >> 4;
    const int bufsz = MT * 16 * APAD;
    const bool wave0 = (tid >> 5) == 0;
#if USE_TDM
    if (wave0)
        tdm_load_tile(Ag + (size_t)m0 * lda, (uint)(uintptr_t)As,
                      MT * 16u, (uint)lda);
#endif
    v16h bfv = load_B_f32(Bg, ldb, 0, col, g);
    for (int k0 = 0; k0 < K; k0 += 32) {
        uint* cur = As + ((k0 >> 5) & 1) * bufsz;
#if USE_TDM
        if (wave0) {
            if (k0 + 32 < K) {
                uint* nxt = As + (((k0 >> 5) + 1) & 1) * bufsz;
                tdm_load_tile(Ag + (size_t)m0 * lda + k0 + 32,
                              (uint)(uintptr_t)nxt, MT * 16u, (uint)lda);
                __builtin_amdgcn_s_wait_tensorcnt(1);
            } else {
                __builtin_amdgcn_s_wait_tensorcnt(0);
            }
        }
#else
        stage_A_manual<MT>(Ag, lda, m0, k0, cur, tid);
#endif
        __syncthreads();
        // Prefetch next-step B while the matrix pipe chews on this step.
        v16h bnext = bfv;
        if (k0 + 32 < K) bnext = load_B_f32(Bg, ldb, k0 + 32, col, g);
        v16h afv[MT];
#pragma unroll
        for (int t = 0; t < MT; ++t) afv[t] = load_A_lds(cur, t, r, g);
#pragma unroll
        for (int t = 0; t < MT; ++t)
            acc[t] = __builtin_amdgcn_wmma_f32_16x16x32_f16(
                false, afv[t], false, bfv, (short)0, acc[t], false, false);
        bfv = bnext;
        __syncthreads();
    }
}

// ---------------------------------------------------------------------------
// 1) Embedding + decaying sinusoidal positional encoding -> tok [N,D] f32
// ---------------------------------------------------------------------------
__global__ __launch_bounds__(256) void embed_pe_kernel(
    const int* __restrict__ ids, const float* __restrict__ emb,
    float* __restrict__ tok) {
    int t = blockIdx.x * 256 + threadIdx.x;
    if (t >= NTOK * DIM) return;
    int d = t % DIM;
    int n = t / DIM;
    int s = n % SEQ;
    const float LN1E4_OVER_D = 9.210340371976184f / (float)DIM;
    float ang = (float)s * __expf(-(float)(d & ~1) * LN1E4_OVER_D);
    float pe = ((d & 1) == 0 ? __sinf(ang) : __cosf(ang)) * __expf(-0.1f * (float)s);
    tok[t] = emb[(size_t)ids[n] * DIM + d] + pe;
}

// ---------------------------------------------------------------------------
// 2) Gate: scores[c,n] = dot(tok[n], gate_w[c]) + gate_b[c]  (wave per output)
// ---------------------------------------------------------------------------
__global__ __launch_bounds__(256) void gate_kernel(
    const float* __restrict__ tok, const float* __restrict__ gate_w,
    const float* __restrict__ gate_b, float* __restrict__ scores) {
    int wid  = threadIdx.x >> 5;
    int lane = threadIdx.x & 31;
    int gw = blockIdx.x * 8 + wid;
    if (gw >= NCOL * NTOK) return;
    int c = gw >> 11;          // / NTOK
    int n = gw & (NTOK - 1);
    const float* tv = tok + (size_t)n * DIM;
    const float* gv = gate_w + (size_t)c * DIM;
    float sum = 0.f;
    for (int t = lane; t < DIM; t += 32) sum += tv[t] * gv[t];
    for (int off = 16; off > 0; off >>= 1) sum += __shfl_xor(sum, off, 32);
    if (lane == 0) scores[c * NTOK + n] = sum + gate_b[c];
}

// ---------------------------------------------------------------------------
// 3) Top-102 per column (block per column, iterative argmax in LDS)
// ---------------------------------------------------------------------------
__global__ __launch_bounds__(256) void topk_kernel(
    const float* __restrict__ scores, int* __restrict__ idx) {
    __shared__ float s[NTOK];
    __shared__ float rv[256];
    __shared__ int   ri[256];
    int c = blockIdx.x, tid = threadIdx.x;
    for (int i = tid; i < NTOK; i += 256) s[i] = scores[c * NTOK + i];
    __syncthreads();
    for (int t = 0; t < TOPK; ++t) {
        float bv = -3.0e38f; int bi = 0x7fffffff;
        for (int i = tid; i < NTOK; i += 256) {
            float v = s[i];
            if (v > bv || (v == bv && i < bi)) { bv = v; bi = i; }
        }
        rv[tid] = bv; ri[tid] = bi;
        __syncthreads();
        for (int off = 128; off > 0; off >>= 1) {
            if (tid < off) {
                float ov = rv[tid + off]; int oi = ri[tid + off];
                if (ov > rv[tid] || (ov == rv[tid] && oi < ri[tid])) {
                    rv[tid] = ov; ri[tid] = oi;
                }
            }
            __syncthreads();
        }
        if (tid == 0) { idx[c * KPAD + t] = ri[0]; s[ri[0]] = -3.0e38f; }
        __syncthreads();
    }
    if (tid < KPAD - TOPK) idx[c * KPAD + TOPK + tid] = -1;
}

// ---------------------------------------------------------------------------
// 4) Gather active rows -> Xa [C,KPAD,D] f16 (padded rows = 0)
// ---------------------------------------------------------------------------
__global__ __launch_bounds__(256) void gather_kernel(
    const float* __restrict__ tok, const int* __restrict__ idx,
    _Float16* __restrict__ Xa) {
    int e = blockIdx.x * 256 + threadIdx.x;
    if (e >= NCOL * KPAD * DIM) return;
    int d  = e % DIM;
    int cj = e / DIM;
    int id = idx[cj];
    Xa[e] = (id >= 0) ? (_Float16)tok[(size_t)id * DIM + d] : (_Float16)0.f;
}

// ---------------------------------------------------------------------------
// 5) MLP layer 1: H = relu(Xa @ W1[c] + b1[c])  (f16 out)
//    grid (6, 1, 16), block 256; block covers all 112 rows x 128 cols.
// ---------------------------------------------------------------------------
__global__ __launch_bounds__(256) void mlp1_kernel(
    const _Float16* __restrict__ Xa, const float* __restrict__ W1,
    const float* __restrict__ b1, _Float16* __restrict__ H) {
    __shared__ uint As[2 * 7 * 16 * APAD];
    int c = blockIdx.z;
    int wid = threadIdx.x >> 5, lane = threadIdx.x & 31;
    int r = lane & 15, g = lane >> 4;
    int col = blockIdx.x * 128 + wid * 16 + r;
    v8f acc[7];
#pragma unroll
    for (int t = 0; t < 7; ++t) acc[t] = (v8f){};
    gemm_f16A<7>(Xa + (size_t)c * KPAD * DIM, DIM, 0,
                 W1 + (size_t)c * DIM * DIM, DIM, col, DIM, As, threadIdx.x, acc);
    float bias = b1[c * DIM + col];
#pragma unroll
    for (int t = 0; t < 7; ++t)
#pragma unroll
        for (int i = 0; i < 8; ++i) {
            float v = fmaxf(acc[t][i] + bias, 0.f);
            H[((size_t)c * KPAD + t * 16 + g * 8 + i) * DIM + col] = (_Float16)v;
        }
}

// ---------------------------------------------------------------------------
// 6) MLP layer 2: Rows = H @ W2[c] + b2[c]  (f16 out)
// ---------------------------------------------------------------------------
__global__ __launch_bounds__(256) void mlp2_kernel(
    const _Float16* __restrict__ H, const float* __restrict__ W2,
    const float* __restrict__ b2, _Float16* __restrict__ Rows) {
    __shared__ uint As[2 * 7 * 16 * APAD];
    int c = blockIdx.z;
    int wid = threadIdx.x >> 5, lane = threadIdx.x & 31;
    int r = lane & 15, g = lane >> 4;
    int col = blockIdx.x * 128 + wid * 16 + r;
    v8f acc[7];
#pragma unroll
    for (int t = 0; t < 7; ++t) acc[t] = (v8f){};
    gemm_f16A<7>(H + (size_t)c * KPAD * DIM, DIM, 0,
                 W2 + (size_t)c * DIM * DIM, DIM, col, DIM, As, threadIdx.x, acc);
    float bias = b2[c * DIM + col];
#pragma unroll
    for (int t = 0; t < 7; ++t)
#pragma unroll
        for (int i = 0; i < 8; ++i)
            Rows[((size_t)c * KPAD + t * 16 + g * 8 + i) * DIM + col] =
                (_Float16)(acc[t][i] + bias);
}

// ---------------------------------------------------------------------------
// 7) mixed init: mixed[n,d] = mixer_b[d]
// ---------------------------------------------------------------------------
__global__ __launch_bounds__(256) void init_mixed_kernel(
    const float* __restrict__ mixer_b, float* __restrict__ mixed) {
    int t = blockIdx.x * 256 + threadIdx.x;
    if (t < NTOK * DIM) mixed[t] = mixer_b[t % DIM];
}

// ---------------------------------------------------------------------------
// 8) Mixer: mixed[idx[c,j], :] += Rows[c,j,:] @ mixer_W[c*D:(c+1)*D, :]
//    (exploits sparsity of col_out -- scatter-add instead of dense [C,N,D])
// ---------------------------------------------------------------------------
__global__ __launch_bounds__(256) void mixer_kernel(
    const _Float16* __restrict__ Rows, const float* __restrict__ mixer_W,
    const int* __restrict__ idx, float* __restrict__ mixed) {
    __shared__ uint As[2 * 7 * 16 * APAD];
    int c = blockIdx.z;
    int wid = threadIdx.x >> 5, lane = threadIdx.x & 31;
    int r = lane & 15, g = lane >> 4;
    int col = blockIdx.x * 128 + wid * 16 + r;
    v8f acc[7];
#pragma unroll
    for (int t = 0; t < 7; ++t) acc[t] = (v8f){};
    gemm_f16A<7>(Rows + (size_t)c * KPAD * DIM, DIM, 0,
                 mixer_W + (size_t)c * DIM * DIM, DIM, col, DIM, As, threadIdx.x, acc);
#pragma unroll
    for (int t = 0; t < 7; ++t)
#pragma unroll
        for (int i = 0; i < 8; ++i) {
            int j = t * 16 + g * 8 + i;
            int tk = idx[c * KPAD + j];
            if (tk >= 0) atomicAdd(&mixed[(size_t)tk * DIM + col], acc[t][i]);
        }
}

// ---------------------------------------------------------------------------
// 8.5) mixed f32 -> f16 (so the logits A-stage is a pure byte copy / TDM-able)
// ---------------------------------------------------------------------------
__global__ __launch_bounds__(256) void convert_mixed_kernel(
    const float* __restrict__ mixed, uint* __restrict__ mixedh) {
    int t = blockIdx.x * 256 + threadIdx.x;
    if (t >= NTOK * DIM / 2) return;
    HPack pk;
    pk.h[0] = (_Float16)mixed[2 * t];
    pk.h[1] = (_Float16)mixed[2 * t + 1];
    mixedh[t] = pk.u32;
}

// ---------------------------------------------------------------------------
// 9) Logits: out = mixedh @ out_W + out_b   [2048,768]x[768,50257]
//    grid (393, 32) block 256; block tile M=64 x N=128; wave holds 4 acc
//    tiles sharing one B fragment per k-step (4x out_W/L2 traffic reduction).
// ---------------------------------------------------------------------------
__global__ __launch_bounds__(256) void logits_kernel(
    const _Float16* __restrict__ mixedh, const float* __restrict__ out_W,
    const float* __restrict__ out_b, float* __restrict__ out) {
    __shared__ uint As[2 * 4 * 16 * APAD];
    int m0 = blockIdx.y * 64;
    int wid = threadIdx.x >> 5, lane = threadIdx.x & 31;
    int r = lane & 15, g = lane >> 4;
    int col = blockIdx.x * 128 + wid * 16 + r;
    int colc = col < VOCAB ? col : VOCAB - 1;
    v8f acc[4];
#pragma unroll
    for (int t = 0; t < 4; ++t) acc[t] = (v8f){};
    gemm_f16A<4>(mixedh, DIM, m0, out_W, VOCAB, colc, DIM, As, threadIdx.x, acc);
    if (col < VOCAB) {
        float bb = out_b[col];
#pragma unroll
        for (int t = 0; t < 4; ++t)
#pragma unroll
            for (int i = 0; i < 8; ++i)
                out[(size_t)(m0 + t * 16 + g * 8 + i) * VOCAB + col] =
                    acc[t][i] + bb;
    }
}

// ---------------------------------------------------------------------------
extern "C" void kernel_launch(void* const* d_in, const int* in_sizes, int n_in,
                              void* d_out, int out_size, void* d_ws, size_t ws_size,
                              hipStream_t stream) {
    const int*   input_ids = (const int*)  d_in[0];
    const float* emb       = (const float*)d_in[1];
    const float* gate_w    = (const float*)d_in[2];
    const float* gate_b    = (const float*)d_in[3];
    const float* W1        = (const float*)d_in[4];
    const float* b1        = (const float*)d_in[5];
    const float* W2        = (const float*)d_in[6];
    const float* b2        = (const float*)d_in[7];
    const float* mixer_W   = (const float*)d_in[8];
    const float* mixer_b   = (const float*)d_in[9];
    const float* out_W     = (const float*)d_in[10];
    const float* out_b     = (const float*)d_in[11];
    float* out = (float*)d_out;

    char* ws = (char*)d_ws;
    float*    tok    = (float*)   (ws);                 //  6291456 B
    float*    scores = (float*)   (ws + 6291456);       //   131072 B
    int*      idxb   = (int*)     (ws + 6422528);       //     7168 B
    _Float16* Xa     = (_Float16*)(ws + 6429696);       //  2752512 B
    _Float16* H      = (_Float16*)(ws + 9182208);       //  2752512 B
    _Float16* Rows   = (_Float16*)(ws + 11934720);      //  2752512 B
    float*    mixed  = (float*)   (ws + 14687232);      //  6291456 B
    _Float16* mixedh = (_Float16*)(ws + 20978688);      //  3145728 B (~24 MB)

    embed_pe_kernel<<<(NTOK * DIM + 255) / 256, 256, 0, stream>>>(input_ids, emb, tok);
    gate_kernel<<<(NCOL * NTOK + 7) / 8, 256, 0, stream>>>(tok, gate_w, gate_b, scores);
    topk_kernel<<<NCOL, 256, 0, stream>>>(scores, idxb);
    gather_kernel<<<(NCOL * KPAD * DIM + 255) / 256, 256, 0, stream>>>(tok, idxb, Xa);

    dim3 gmlp(DIM / 128, 1, NCOL); // (6,1,16)
    mlp1_kernel<<<gmlp, 256, 0, stream>>>(Xa, W1, b1, H);
    mlp2_kernel<<<gmlp, 256, 0, stream>>>(H, W2, b2, Rows);

    init_mixed_kernel<<<(NTOK * DIM + 255) / 256, 256, 0, stream>>>(mixer_b, mixed);
    mixer_kernel<<<gmlp, 256, 0, stream>>>(Rows, mixer_W, idxb, mixed);
    convert_mixed_kernel<<<(NTOK * DIM / 2 + 255) / 256, 256, 0, stream>>>(
        mixed, (uint*)mixedh);

    dim3 glog((VOCAB + 127) / 128, NTOK / 64); // (393,32)
    logits_kernel<<<glog, 256, 0, stream>>>(mixedh, out_W, out_b, out);
}
